// HSTGAttn_7172595384476
// MI455X (gfx1250) — matrix operand order repaired and checked by reading, not compiled, not run on hardware
//
#include <hip/hip_runtime.h>

typedef __attribute__((ext_vector_type(16))) __bf16 v16bf;
typedef __attribute__((ext_vector_type(16))) float  v16f;
typedef __attribute__((ext_vector_type(8)))  float  v8f;

constexpr int NSN = 50000;
constexpr int NTN = 4096;
constexpr int CH  = 64;

// ---------- helpers ----------

// hardware float->bf16 (vector fptrunc -> packed v_cvt_pk_bf16_f32)
__device__ __forceinline__ v16bf cvt16(v16f f) {
  return __builtin_convertvector(f, v16bf);
}
__device__ __forceinline__ __bf16 f2bf(float f) { return (__bf16)f; }

// order-preserving float<->uint for atomicMax-based segment max
__device__ __forceinline__ unsigned fenc(float f) {
  unsigned u = __builtin_bit_cast(unsigned, f);
  return (u & 0x80000000u) ? ~u : (u | 0x80000000u);
}
__device__ __forceinline__ float fdec(unsigned u) {
  unsigned b = (u & 0x80000000u) ? (u & 0x7FFFFFFFu) : ~u;
  return __builtin_bit_cast(float, b);
}

// K offset of A-fragment element pair vv (elements 2vv,2vv+1) for lane half
__device__ __forceinline__ int a_kpair(int vv, int half) {
  return ((vv >> 2) << 4) + half * 8 + ((vv & 3) << 1);
}

// Pre-swizzle a (KT*32) x 64 weight matrix into LDS so that each
// (kt, nt, lane) fragment is 16 contiguous bf16 (one v16bf load = 2x ds_load_b128).
// Fragment element e of lane ln maps to k = kt*32 + (e>>3)*16 + (ln>>4)*8 + (e&7), n = nt*16 + (ln&15).
__device__ __forceinline__ void load_b_swz(const float* __restrict__ W, __bf16* Bs,
                                           int kts, int tid, int nthreads) {
  const int total = kts * 2048;            // kts * 32k * 64n
  for (int i = tid; i < total; i += nthreads) {
    const int e  = i & 15;
    const int ln = (i >> 4) & 31;
    const int nt = (i >> 9) & 3;
    const int kt = i >> 11;
    const int k = kt * 32 + ((e >> 3) << 4) + ((ln >> 4) << 3) + (e & 7);
    const int n = nt * 16 + (ln & 15);
    Bs[i] = f2bf(W[k * CH + n]);
  }
}

// ---------- small kernels ----------

__global__ void zero_kernel(float* p, size_t n) {
  size_t i = (size_t)blockIdx.x * blockDim.x + threadIdx.x;
  size_t s = (size_t)gridDim.x * blockDim.x;
  for (; i < n; i += s) p[i] = 0.f;
}

__global__ void dot_kernel(const float* __restrict__ x, const float* __restrict__ v,
                           float* __restrict__ out, int N) {
  int i = blockIdx.x * blockDim.x + threadIdx.x;
  if (i >= N) return;
  float s = 0.f;
#pragma unroll
  for (int c = 0; c < CH; ++c) s += x[(size_t)i * CH + c] * v[c];
  out[i] = s;
}

__global__ void alpha_max_kernel(const float* __restrict__ aS, const float* __restrict__ aD,
                                 const int* __restrict__ src, const int* __restrict__ dst,
                                 unsigned* __restrict__ menc, int E) {
  int i = blockIdx.x * blockDim.x + threadIdx.x;
  if (i >= E) return;
  float a = aS[src[i]] + aD[dst[i]];
  a = (a >= 0.f) ? a : 0.2f * a;
  atomicMax(&menc[dst[i]], fenc(a));
}

__global__ void exp_sum_kernel(const float* __restrict__ aS, const float* __restrict__ aD,
                               const int* __restrict__ src, const int* __restrict__ dst,
                               const unsigned* __restrict__ menc, float* __restrict__ ssum,
                               float* __restrict__ ebuf, int E) {
  int i = blockIdx.x * blockDim.x + threadIdx.x;
  if (i >= E) return;
  float a = aS[src[i]] + aD[dst[i]];
  a = (a >= 0.f) ? a : 0.2f * a;
  float e = expf(a - fdec(menc[dst[i]]));
  ebuf[i] = e;
  atomicAdd(&ssum[dst[i]], e);
}

// ebuf[i] <- normalized attention coefficient
__global__ void scale_kernel(const int* __restrict__ dst, const float* __restrict__ ssum,
                             float* __restrict__ ebuf, int E) {
  int i = blockIdx.x * blockDim.x + threadIdx.x;
  if (i >= E) return;
  ebuf[i] = ebuf[i] / (ssum[dst[i]] + 1e-16f);
}

// u = ep_w @ W3 ; c0 = ep_b @ W3 + att_b   (W3 = att_w rows 128..191)
__global__ void epvec_kernel(const float* __restrict__ att_w, const float* __restrict__ att_b,
                             const float* __restrict__ ep_w,  const float* __restrict__ ep_b,
                             float* __restrict__ u, float* __restrict__ c0) {
  int t = threadIdx.x;          // blockDim = 128
  int n = t & 63;
  if (t < 64) {
    float s = 0.f;
#pragma unroll
    for (int k = 0; k < CH; ++k) s += ep_w[k] * att_w[(128 + k) * CH + n];
    u[n] = s;
  } else {
    float s = att_b[n];
#pragma unroll
    for (int k = 0; k < CH; ++k) s += ep_b[k] * att_w[(128 + k) * CH + n];
    c0[n] = s;
  }
}

// ---------- generic WMMA GEMM: out = X @ Wm  (N x 64 @ 64 x 64) ----------

__global__ __launch_bounds__(128) void gemm64_kernel(
    const float* __restrict__ X, const float* __restrict__ Wm,
    float* __restrict__ out, int N)
{
  __shared__ __align__(32) __bf16 Bs[CH * CH];
  load_b_swz(Wm, Bs, 2, threadIdx.x, blockDim.x);
  __syncthreads();
  const v16bf* Bsw = reinterpret_cast<const v16bf*>(Bs);

  const int wave = threadIdx.x >> 5, lane = threadIdx.x & 31;
  const int m16 = lane & 15, half = lane >> 4;
  const int ntile = N >> 4;
  for (int tile = blockIdx.x * 4 + wave; tile < ntile; tile += gridDim.x * 4) {
    const int row = (tile << 4) + m16;
    v8f acc[4];
#pragma unroll
    for (int nt = 0; nt < 4; ++nt)
#pragma unroll
      for (int v = 0; v < 8; ++v) acc[nt][v] = 0.f;
#pragma unroll
    for (int kt = 0; kt < 2; ++kt) {
      v16f af;
#pragma unroll
      for (int vv = 0; vv < 8; ++vv) {
        const float2 f2 = *(const float2*)(X + (size_t)row * CH + kt * 32 + a_kpair(vv, half));
        af[2 * vv] = f2.x; af[2 * vv + 1] = f2.y;
      }
      const v16bf a = cvt16(af);
#pragma unroll
      for (int nt = 0; nt < 4; ++nt) {
        const v16bf b = Bsw[(kt * 4 + nt) * 32 + lane];
        acc[nt] = __builtin_amdgcn_wmma_f32_16x16x32_bf16(
            false, a, false, b, (short)0, acc[nt], false, false);
      }
    }
    // one base address per row tile; row and n-tile offsets are unroll-constant (fold into store imm)
    float* ob = out + (size_t)((tile << 4) + 8 * half) * CH + m16;
#pragma unroll
    for (int nt = 0; nt < 4; ++nt)
#pragma unroll
      for (int v = 0; v < 8; ++v)
        ob[v * CH + nt * 16] = acc[nt][v];
  }
}

// ---------- streaming per-edge pass ----------
// att = P1[src] + P2[dst] + wE*u + c0 ; msg = xj * alpha * att ; aggr[dst] += msg
// 16 edges per 256-thread block; per-edge scalars + u/c0 staged in LDS.

__global__ __launch_bounds__(256) void edge_msg_kernel(
    const float* __restrict__ P1, const float* __restrict__ P2,
    const float* __restrict__ x_src,
    const int* __restrict__ src, const int* __restrict__ dst,
    const float* __restrict__ ew, const float* __restrict__ scl,
    const float* __restrict__ u, const float* __restrict__ c0,
    float* __restrict__ aggr, int E)
{
  __shared__ int   sS[16], sD[16];
  __shared__ float sW[16], sC[16];
  __shared__ __align__(16) float sUC[128];   // [0:64) = u, [64:128) = c0
  const int t  = threadIdx.x;
  const int eb = blockIdx.x * 16;
  if (t < 16) {
    sS[t] = src[eb + t];
    sD[t] = dst[eb + t];
    sW[t] = ew[eb + t];
    sC[t] = scl[eb + t];
  } else if (t < 144) {
    const int i = t - 16;
    sUC[i] = (i < 64) ? u[i] : c0[i - 64];
  }
  __syncthreads();
  const int le = t >> 4;          // local edge 0..15
  const int q  = (t & 15) * 4;    // channel base
  const unsigned sN = (unsigned)sS[le];
  const unsigned dN = (unsigned)sD[le];
  const float wE = sW[le];
  const float sc = sC[le];
  const float4 p1 = *(const float4*)(P1 + (size_t)sN * CH + q);
  const float4 p2 = *(const float4*)(P2 + (size_t)dN * CH + q);
  const float4 xj = *(const float4*)(x_src + (size_t)sN * CH + q);
  const float4 uu = *(const float4*)(&sUC[q]);
  const float4 cc = *(const float4*)(&sUC[64 + q]);
  float* ag = aggr + (size_t)dN * CH + q;
  atomicAdd(ag + 0, xj.x * sc * (p1.x + p2.x + wE * uu.x + cc.x));
  atomicAdd(ag + 1, xj.y * sc * (p1.y + p2.y + wE * uu.y + cc.y));
  atomicAdd(ag + 2, xj.z * sc * (p1.z + p2.z + wE * uu.z + cc.z));
  atomicAdd(ag + 3, xj.w * sc * (p1.w + p2.w + wE * uu.w + cc.w));
}

// ---------- per-node WMMA kernel: out = relu([aggr|x] @ agg_w + agg_b) ----------

__global__ __launch_bounds__(128) void node_agg_kernel(
    const float* __restrict__ aggr, const float* __restrict__ xd,
    const float* __restrict__ agg_w, const float* __restrict__ agg_b,
    float* __restrict__ out, int N)
{
  __shared__ __align__(32) __bf16 Bs[128 * CH];
  load_b_swz(agg_w, Bs, 4, threadIdx.x, blockDim.x);
  __syncthreads();
  const v16bf* Bsw = reinterpret_cast<const v16bf*>(Bs);

  const int wave = threadIdx.x >> 5, lane = threadIdx.x & 31;
  const int m16 = lane & 15, half = lane >> 4;
  float bias[4];
#pragma unroll
  for (int nt = 0; nt < 4; ++nt) bias[nt] = agg_b[nt * 16 + m16];
  const int ntile = N >> 4;
  for (int tile = blockIdx.x * 4 + wave; tile < ntile; tile += gridDim.x * 4) {
    const int row = (tile << 4) + m16;
    v8f acc[4];
#pragma unroll
    for (int nt = 0; nt < 4; ++nt)
#pragma unroll
      for (int v = 0; v < 8; ++v) acc[nt][v] = bias[nt];
#pragma unroll
    for (int kt = 0; kt < 4; ++kt) {                    // K = 128
      const float* srcp = (kt < 2) ? (aggr + (size_t)row * CH + kt * 32)
                                   : (xd + (size_t)row * CH + (kt - 2) * 32);
      v16f af;
#pragma unroll
      for (int vv = 0; vv < 8; ++vv) {
        const float2 f2 = *(const float2*)(srcp + a_kpair(vv, half));
        af[2 * vv] = f2.x; af[2 * vv + 1] = f2.y;
      }
      const v16bf a = cvt16(af);
#pragma unroll
      for (int nt = 0; nt < 4; ++nt) {
        const v16bf b = Bsw[(kt * 4 + nt) * 32 + lane];
        acc[nt] = __builtin_amdgcn_wmma_f32_16x16x32_bf16(
            false, a, false, b, (short)0, acc[nt], false, false);
      }
    }
    float* ob = out + (size_t)((tile << 4) + 8 * half) * CH + m16;
#pragma unroll
    for (int nt = 0; nt < 4; ++nt)
#pragma unroll
      for (int v = 0; v < 8; ++v)
        ob[v * CH + nt * 16] = fmaxf(acc[nt][v], 0.f);
  }
}

// ---------- group kernels: kpart += sum_rows tanh(o @ k_w + k_b) ----------

__global__ __launch_bounds__(128) void kmean_kernel(
    const float* __restrict__ o, const float* __restrict__ kw,
    const float* __restrict__ kb, float* __restrict__ kpart, int N)
{
  __shared__ __align__(32) __bf16 Bs[CH * CH];
  load_b_swz(kw, Bs, 2, threadIdx.x, blockDim.x);
  __syncthreads();
  const v16bf* Bsw = reinterpret_cast<const v16bf*>(Bs);

  const int wave = threadIdx.x >> 5, lane = threadIdx.x & 31;
  const int m16 = lane & 15, half = lane >> 4;
  float bias[4];
#pragma unroll
  for (int nt = 0; nt < 4; ++nt) bias[nt] = kb[nt * 16 + m16];
  const int ntile = N >> 4;
  for (int tile = blockIdx.x * 4 + wave; tile < ntile; tile += gridDim.x * 4) {
    const int row = (tile << 4) + m16;
    v8f acc[4];
#pragma unroll
    for (int nt = 0; nt < 4; ++nt)
#pragma unroll
      for (int v = 0; v < 8; ++v) acc[nt][v] = bias[nt];
#pragma unroll
    for (int kt = 0; kt < 2; ++kt) {                    // K = 64
      v16f af;
#pragma unroll
      for (int vv = 0; vv < 8; ++vv) {
        const float2 f2 = *(const float2*)(o + (size_t)row * CH + kt * 32 + a_kpair(vv, half));
        af[2 * vv] = f2.x; af[2 * vv + 1] = f2.y;
      }
      const v16bf a = cvt16(af);
#pragma unroll
      for (int nt = 0; nt < 4; ++nt) {
        const v16bf b = Bsw[(kt * 4 + nt) * 32 + lane];
        acc[nt] = __builtin_amdgcn_wmma_f32_16x16x32_bf16(
            false, a, false, b, (short)0, acc[nt], false, false);
      }
    }
#pragma unroll
    for (int nt = 0; nt < 4; ++nt) {
      float s = 0.f;
#pragma unroll
      for (int v = 0; v < 8; ++v) s += tanhf(acc[nt][v]);
      atomicAdd(&kpart[nt * 16 + m16], s);
    }
  }
}

__global__ void attn_kernel(const float* __restrict__ kpart,
                            const float* __restrict__ q_s, const float* __restrict__ q_t,
                            float* __restrict__ attn)
{
  if (threadIdx.x != 0 || blockIdx.x != 0) return;
  float s0 = 0.f, s1 = 0.f, t0 = 0.f, t1 = 0.f;
  for (int c = 0; c < CH; ++c) {
    s0 += q_s[c] * kpart[c]        * (1.f / NSN);
    s1 += q_s[c] * kpart[64 + c]   * (1.f / NSN);
    t0 += q_t[c] * kpart[128 + c]  * (1.f / NTN);
    t1 += q_t[c] * kpart[192 + c]  * (1.f / NTN);
  }
  float ms = fmaxf(s0, s1), e0 = expf(s0 - ms), e1 = expf(s1 - ms);
  attn[0] = e0 / (e0 + e1); attn[1] = e1 / (e0 + e1);
  float mt = fmaxf(t0, t1), f0 = expf(t0 - mt), f1 = expf(t1 - mt);
  attn[2] = f0 / (f0 + f1); attn[3] = f1 / (f0 + f1);
}

__global__ void combine_kernel(const float* __restrict__ o_ss, const float* __restrict__ o_ts,
                               const float* __restrict__ o_tt, const float* __restrict__ o_st,
                               const float* __restrict__ attn, float* __restrict__ out)
{
  size_t i = (size_t)blockIdx.x * blockDim.x + threadIdx.x;
  const size_t nS = (size_t)NSN * CH, nT = (size_t)NTN * CH;
  if (i < nS) out[i]      = attn[0] * o_ss[i] + attn[1] * o_ts[i];
  if (i < nT) out[nS + i] = attn[2] * o_tt[i] + attn[3] * o_st[i];
}

// ---------- host ----------

extern "C" void kernel_launch(void* const* d_in, const int* in_sizes, int n_in,
                              void* d_out, int out_size, void* d_ws, size_t ws_size,
                              hipStream_t stream)
{
  (void)n_in; (void)out_size; (void)ws_size;
  const float* x      = (const float*)d_in[0];
  const int*   src_ss = (const int*)d_in[1];
  const int*   dst_ss = (const int*)d_in[2];
  const float* w_ss   = (const float*)d_in[3];
  const int*   src_tt = (const int*)d_in[4];
  const int*   dst_tt = (const int*)d_in[5];
  const float* w_tt   = (const float*)d_in[6];
  const int*   src_st = (const int*)d_in[7];
  const int*   dst_st = (const int*)d_in[8];
  const float* w_st   = (const float*)d_in[9];
  const int*   src_ts = (const int*)d_in[10];
  const int*   dst_ts = (const int*)d_in[11];
  const float* w_ts   = (const float*)d_in[12];
  const int E_ss = in_sizes[1], E_tt = in_sizes[4], E_st = in_sizes[7], E_ts = in_sizes[10];

  // params flattened pytree order (sorted keys):
  // node_s{k_b,k_w,q}=13..15, node_t=16..18,
  // then ss=19, st=27, ts=35, tt=43 each {agg_b,agg_w,att_b,att_w,ep_b,ep_w,lin_dst,lin_src}
  const float* ns_kb = (const float*)d_in[13];
  const float* ns_kw = (const float*)d_in[14];
  const float* ns_q  = (const float*)d_in[15];
  const float* nt_kb = (const float*)d_in[16];
  const float* nt_kw = (const float*)d_in[17];
  const float* nt_q  = (const float*)d_in[18];
  auto P = [&](int base, int k) { return (const float*)d_in[base + k]; };
  // k: 0=agg_b 1=agg_w 2=att_b 3=att_w 4=ep_b 5=ep_w 6=lin_dst 7=lin_src

  // workspace layout (floats); zero-init region first
  float* W = (float*)d_ws;
  size_t off = 0;
  auto take = [&](size_t n) { float* p = W + off; off += n; return p; };
  float* m_ss = take(NSN); float* m_ts = take(NSN);
  float* m_tt = take(NTN); float* m_st = take(NTN);
  float* s_ss = take(NSN); float* s_ts = take(NSN);
  float* s_tt = take(NTN); float* s_st = take(NTN);
  float* ag_ss = take((size_t)NSN * CH); float* ag_ts = take((size_t)NSN * CH);
  float* ag_tt = take((size_t)NTN * CH); float* ag_st = take((size_t)NTN * CH);
  float* kpart = take(256);
  float* attn  = take(4);
  const size_t zeroN = off;
  float* aS_ss = take(NSN); float* aD_ss = take(NSN);
  float* aS_tt = take(NTN); float* aD_tt = take(NTN);
  float* aS_st = take(NSN); float* aD_st = take(NTN);
  float* aS_ts = take(NTN); float* aD_ts = take(NSN);
  float* eb_ss = take(E_ss); float* eb_tt = take(E_tt);
  float* eb_st = take(E_st); float* eb_ts = take(E_ts);
  float* o_ss = take((size_t)NSN * CH); float* o_ts = take((size_t)NSN * CH);
  float* o_tt = take((size_t)NTN * CH); float* o_st = take((size_t)NTN * CH);
  float* P1buf = take((size_t)NSN * CH);   // reused across edge types
  float* P2buf = take((size_t)NSN * CH);   // reused across edge types
  float* uvec  = take(CH);
  float* c0vec = take(CH);

  zero_kernel<<<2048, 256, 0, stream>>>(W, zeroN);

  const float* xs = x;
  const float* xt = x + (size_t)NSN * CH;
  auto dotl = [&](const float* xx, const float* v, float* out, int N) {
    dot_kernel<<<(N + 255) / 256, 256, 0, stream>>>(xx, v, out, N);
  };
  dotl(xs, P(19, 7), aS_ss, NSN);  dotl(xs, P(19, 6), aD_ss, NSN);
  dotl(xt, P(43, 7), aS_tt, NTN);  dotl(xt, P(43, 6), aD_tt, NTN);
  dotl(xs, P(27, 7), aS_st, NSN);  dotl(xt, P(27, 6), aD_st, NTN);
  dotl(xt, P(35, 7), aS_ts, NTN);  dotl(xs, P(35, 6), aD_ts, NSN);

  struct EdgeJob {
    const float* xsrc; const float* xdst; int nsrc; int ndst;
    const int* src; const int* dst; const float* w; int E; int pb;
    float* aS; float* aD; float* m; float* s; float* eb; float* ag; float* out;
  };
  EdgeJob jobs[4] = {
    { xs, xs, NSN, NSN, src_ss, dst_ss, w_ss, E_ss, 19, aS_ss, aD_ss, m_ss, s_ss, eb_ss, ag_ss, o_ss },
    { xt, xt, NTN, NTN, src_tt, dst_tt, w_tt, E_tt, 43, aS_tt, aD_tt, m_tt, s_tt, eb_tt, ag_tt, o_tt },
    { xs, xt, NSN, NTN, src_st, dst_st, w_st, E_st, 27, aS_st, aD_st, m_st, s_st, eb_st, ag_st, o_st },
    { xt, xs, NTN, NSN, src_ts, dst_ts, w_ts, E_ts, 35, aS_ts, aD_ts, m_ts, s_ts, eb_ts, ag_ts, o_ts },
  };

  auto gemm_grid = [](int N) { int g = (N / 16 + 3) / 4; return g > 2048 ? 2048 : g; };

  for (int j = 0; j < 4; ++j) {
    EdgeJob& e = jobs[j];
    const int gE = (e.E + 255) / 256;
    alpha_max_kernel<<<gE, 256, 0, stream>>>(e.aS, e.aD, e.src, e.dst, (unsigned*)e.m, e.E);
    exp_sum_kernel<<<gE, 256, 0, stream>>>(e.aS, e.aD, e.src, e.dst,
                                           (const unsigned*)e.m, e.s, e.eb, e.E);
    scale_kernel<<<gE, 256, 0, stream>>>(e.dst, e.s, e.eb, e.E);

    const float* attw = P(e.pb, 3);
    // P1 = X_src @ W1, P2 = X_dst @ W2 (per-node, not per-edge)
    gemm64_kernel<<<gemm_grid(e.nsrc), 128, 0, stream>>>(e.xsrc, attw, P1buf, e.nsrc);
    gemm64_kernel<<<gemm_grid(e.ndst), 128, 0, stream>>>(e.xdst, attw + 64 * CH, P2buf, e.ndst);
    epvec_kernel<<<1, 128, 0, stream>>>(attw, P(e.pb, 2), P(e.pb, 5), P(e.pb, 4), uvec, c0vec);

    edge_msg_kernel<<<e.E / 16, 256, 0, stream>>>(P1buf, P2buf, e.xsrc, e.src, e.dst, e.w, e.eb,
                                                  uvec, c0vec, e.ag, e.E);
  }

  for (int j = 0; j < 4; ++j) {
    EdgeJob& e = jobs[j];
    node_agg_kernel<<<gemm_grid(e.ndst), 128, 0, stream>>>(
        e.ag, e.xdst, P(e.pb, 1), P(e.pb, 0), e.out, e.ndst);
  }

  auto km = [&](const float* o, const float* kw, const float* kb, float* kp, int N) {
    kmean_kernel<<<gemm_grid(N), 128, 0, stream>>>(o, kw, kb, kp, N);
  };
  km(o_ss, ns_kw, ns_kb, kpart + 0,   NSN);
  km(o_ts, ns_kw, ns_kb, kpart + 64,  NSN);
  km(o_tt, nt_kw, nt_kb, kpart + 128, NTN);
  km(o_st, nt_kw, nt_kb, kpart + 192, NTN);

  attn_kernel<<<1, 1, 0, stream>>>(kpart, ns_q, nt_q, attn);

  size_t nS = (size_t)NSN * CH;
  combine_kernel<<<(unsigned)((nS + 255) / 256), 256, 0, stream>>>(
      o_ss, o_ts, o_tt, o_st, attn, (float*)d_out);
}
